// Net_89111981457831
// MI455X (gfx1250) — matrix operand order repaired and checked by reading, not compiled, not run on hardware
//
#include <hip/hip_runtime.h>
#include <cstdint>
#include <cstddef>

typedef __attribute__((ext_vector_type(16))) __bf16 v16bf;
typedef __attribute__((ext_vector_type(8)))  float  v8f;

#define TSTEPS 100

__device__ __forceinline__ float sigm(float x) { return 1.0f / (1.0f + __expf(-x)); }

// ---------------------------------------------------------------------------
// WMMA fragment helpers (CDNA5 wave32 layouts, 16x16x32 bf16)
// A: 16xK bf16, row-major in LDS with row stride `strideA` (elements).
//    lane L: row = L%16, grp = L/16; elem e -> K=kc*32+grp*16+e
//    (16 contiguous K).
// B: KxN weight pre-transposed to Bt[N][Kpad] bf16 in global memory.
//    lane L: N = n0 + L%16, Kbase = kc*32 + (L/16)*16, 16 contiguous K.
// C/D: lane L: N = n0 + L%16; vgpr r: M = r + (L/16)*8.
// ---------------------------------------------------------------------------
__device__ __forceinline__ v16bf load_A(const __bf16* A, int strideA, int kc, int lane) {
    int row = lane & 15;
    int grp = lane >> 4;
    const __bf16* p = A + row * strideA + kc * 32 + grp * 16;
    v16bf a;
#pragma unroll
    for (int e = 0; e < 16; ++e) a[e] = p[e];
    return a;
}

__device__ __forceinline__ v16bf load_B(const __bf16* Bt, int Kpad, int n0, int kc, int lane) {
    int n  = n0 + (lane & 15);
    int kb = kc * 32 + (lane >> 4) * 16;
    const __bf16* p = Bt + (size_t)n * Kpad + kb;
    v16bf b;
#pragma unroll
    for (int e = 0; e < 16; ++e) b[e] = p[e];
    return b;
}

// Two adjacent 16x16 output tiles (cols n0 and n0+16) sharing one A fragment.
__device__ __forceinline__ void gemm_tile2(const __bf16* A, int strideA,
                                           const __bf16* Bt, int Kpad, int n0,
                                           int kChunks, int lane,
                                           v8f& acc0, v8f& acc1) {
#pragma unroll 2
    for (int kc = 0; kc < kChunks; ++kc) {
        v16bf a  = load_A(A, strideA, kc, lane);
        v16bf b0 = load_B(Bt, Kpad, n0, kc, lane);
        v16bf b1 = load_B(Bt, Kpad, n0 + 16, kc, lane);
        acc0 = __builtin_amdgcn_wmma_f32_16x16x32_bf16(false, a, false, b0,
                                                       (short)0, acc0, false, false);
        acc1 = __builtin_amdgcn_wmma_f32_16x16x32_bf16(false, a, false, b1,
                                                       (short)0, acc1, false, false);
    }
}

__device__ __forceinline__ v8f gemm_tile(const __bf16* A, int strideA,
                                         const __bf16* Bt, int Kpad,
                                         int n0, int kChunks, int lane) {
    v8f acc = {0.f, 0.f, 0.f, 0.f, 0.f, 0.f, 0.f, 0.f};
#pragma unroll 2
    for (int kc = 0; kc < kChunks; ++kc) {
        v16bf a = load_A(A, strideA, kc, lane);
        v16bf b = load_B(Bt, Kpad, n0, kc, lane);
        acc = __builtin_amdgcn_wmma_f32_16x16x32_bf16(false, a, false, b,
                                                      (short)0, acc, false, false);
    }
    return acc;
}

__device__ __forceinline__ void store_C(float* C, int strideC, int n0, v8f acc, int lane) {
    int col   = n0 + (lane & 15);
    int rbase = (lane >> 4) * 8;
#pragma unroll
    for (int r = 0; r < 8; ++r) C[(rbase + r) * strideC + col] = acc[r];
}

// ---------------------------------------------------------------------------
// Parameter bundle for the persistent kernel
// ---------------------------------------------------------------------------
struct KParams {
    const float *x_l, *x_v, *x_a;
    const float *gvW, *gvB, *gaW, *gaB;
    const float *bL, *bA, *bV;                 // LSTHM biases (512 each)
    const float *bl1, *bv1, *bl2, *bv2;        // cross-block biases (128 each)
    const float *zb1, *zb2, *gb1, *gb2;        // z_hat / gamma biases
    const float *pb1, *pW2, *pb2;              // prediction head
    const __bf16 *WcatL, *WcatA, *WcatV;       // [512][576/352/320]
    const __bf16 *Wk1t, *Wk2t;                 // 4 x [128][128]
    const __bf16 *Wv2l1t, *Wl2v1t, *Wv2l2t, *Wl2v2t;  // [128][128]
    const __bf16 *zW1t, *zW2t, *gW1t, *gW2t;   // [256][384], [128][256]
    const __bf16 *pW1t;                        // [128][128]
    float* out;
};

__global__ __launch_bounds__(256) void fuse_kernel(KParams P) {
    // A buffers hold [x | pad | h | z] rows in bf16 (WMMA A operand source)
    __shared__ __align__(16) __bf16 Al[16 * 576];   // x:0..299 h:320 z:448
    __shared__ __align__(16) __bf16 Aa[16 * 352];   // x:0..73  h:96  z:224
    __shared__ __align__(16) __bf16 Av[16 * 320];   // x:0..46  h:64  z:192
    __shared__ __align__(16) float  Gb[3][16 * 512];   // GEMM outputs / scratch
    __shared__ __align__(16) float  Cst[3][16 * 128];  // cell states (l,a,v)
    __shared__ __align__(16) float  Hf[3][16 * 128];   // fp32 hidden (l,a,v)
    __shared__ __align__(16) float  Xraw[16 * 121];    // raw xa(74) | xv(47)
    __shared__ __align__(16) __bf16 Hcat[16 * 384];
    __shared__ __align__(16) __bf16 Za[2][16 * 256];
    __shared__ float Gg2[128];                          // per-k tanh terms
    __shared__ float Gg[32];                            // cross gates

    const int tid  = threadIdx.x;
    const int lane = tid & 31;
    const int uwv  = __builtin_amdgcn_readfirstlane(tid >> 5);  // scalar wave id
    const int r0   = blockIdx.x * 16;

    for (int i = tid; i < 16 * 576; i += 256) Al[i] = (__bf16)0.0f;
    for (int i = tid; i < 16 * 352; i += 256) Aa[i] = (__bf16)0.0f;
    for (int i = tid; i < 16 * 320; i += 256) Av[i] = (__bf16)0.0f;
    for (int i = tid; i < 3 * 16 * 128; i += 256) (&Cst[0][0])[i] = 0.0f;
    __syncthreads();

    for (int t = 0; t < TSTEPS; ++t) {
        // ---- phase 1: stage inputs + prefetch next step ----------------
        if (t + 1 < TSTEPS) {
            if (tid < 16)
                __builtin_prefetch(&P.x_l[((size_t)(r0 + tid) * TSTEPS + t + 1) * 300], 0, 3);
            else if (tid < 32)
                __builtin_prefetch(&P.x_a[((size_t)(r0 + tid - 16) * TSTEPS + t + 1) * 74], 0, 3);
            else if (tid < 48)
                __builtin_prefetch(&P.x_v[((size_t)(r0 + tid - 32) * TSTEPS + t + 1) * 47], 0, 3);
        }
        for (int i = tid; i < 16 * 300; i += 256) {
            int r = i / 300, d = i - r * 300;
            Al[r * 576 + d] = (__bf16)P.x_l[((size_t)(r0 + r) * TSTEPS + t) * 300 + d];
        }
        for (int i = tid; i < 16 * 74; i += 256) {
            int r = i / 74, d = i - r * 74;
            Xraw[r * 121 + d] = P.x_a[((size_t)(r0 + r) * TSTEPS + t) * 74 + d];
        }
        for (int i = tid; i < 16 * 47; i += 256) {
            int r = i / 47, d = i - r * 47;
            Xraw[r * 121 + 74 + d] = P.x_v[((size_t)(r0 + r) * TSTEPS + t) * 47 + d];
        }
        __syncthreads();
        // ---- phase 1b: input sigmoid gates (small, VALU) ---------------
        for (int i = tid; i < 16 * 74; i += 256) {
            int r = i / 74, j = i - r * 74;
            float s = P.gaB[j];
            const float* xr = &Xraw[r * 121];
            for (int k = 0; k < 74; ++k) s += xr[k] * P.gaW[k * 74 + j];
            Aa[r * 352 + j] = (__bf16)(sigm(s) * xr[j]);
        }
        for (int i = tid; i < 16 * 47; i += 256) {
            int r = i / 47, j = i - r * 47;
            float s = P.gvB[j];
            const float* xr = &Xraw[r * 121 + 74];
            for (int k = 0; k < 47; ++k) s += xr[k] * P.gvW[k * 47 + j];
            Av[r * 320 + j] = (__bf16)(sigm(s) * xr[j]);
        }
        __syncthreads();
        // ---- phase 2: LSTHM gate GEMMs [x|h|z] @ Wcat -> 16x512 --------
        // 48 jobs, each = 2 adjacent N-tiles (32 cols)
        for (int job = uwv; job < 48; job += 8) {
            int mod = job >> 4, nt2 = job & 15;
            const __bf16 *A, *W; const float* bias; int st, kc;
            if (mod == 0)      { A = Al; st = 576; kc = 18; W = P.WcatL; bias = P.bL; }
            else if (mod == 1) { A = Aa; st = 352; kc = 11; W = P.WcatA; bias = P.bA; }
            else               { A = Av; st = 320; kc = 10; W = P.WcatV; bias = P.bV; }
            v8f a0 = {0.f,0.f,0.f,0.f,0.f,0.f,0.f,0.f};
            v8f a1 = {0.f,0.f,0.f,0.f,0.f,0.f,0.f,0.f};
            gemm_tile2(A, st, W, st, nt2 * 32, kc, lane, a0, a1);
            float b0 = bias[nt2 * 32 + (lane & 15)];
            float b1 = bias[nt2 * 32 + 16 + (lane & 15)];
#pragma unroll
            for (int r = 0; r < 8; ++r) { a0[r] += b0; a1[r] += b1; }
            store_C(&Gb[mod][0], 512, nt2 * 32,      a0, lane);
            store_C(&Gb[mod][0], 512, nt2 * 32 + 16, a1, lane);
        }
        __syncthreads();
        // ---- phase 3: LSTHM nonlinearity -------------------------------
        for (int i = tid; i < 3 * 2048; i += 256) {
            int mod = i >> 11, rem = i & 2047, r = rem >> 7, j = rem & 127;
            float gi = Gb[mod][r * 512 + j];
            float gf = Gb[mod][r * 512 + 128 + j];
            float gg = Gb[mod][r * 512 + 256 + j];
            float go = Gb[mod][r * 512 + 384 + j];
            float c = sigm(gf) * Cst[mod][rem] + sigm(gi) * tanhf(gg);
            Cst[mod][rem] = c;
            float h = sigm(go) * tanhf(c);
            Hf[mod][rem] = h;
            if (mod == 0)      Al[r * 576 + 320 + j] = (__bf16)h;
            else if (mod == 1) Aa[r * 352 + 96  + j] = (__bf16)h;
            else               Av[r * 320 + 64  + j] = (__bf16)h;
        }
        __syncthreads();
        // ---- phase 4: cross GEMMs (bilinear u_k and projections) -------
        // 48 jobs of 2 tiles: 32 bilinear (2 cross x 4 k x 4), 16 proj (4 gemm x 4)
        for (int job = uwv; job < 48; job += 8) {
            if (job < 32) {
                int cross = job >> 4, rest = job & 15, k = rest >> 2, nt2 = rest & 3;
                const __bf16* Wt = (cross ? P.Wk2t : P.Wk1t) + (size_t)k * 16384;
                v8f a0 = {0.f,0.f,0.f,0.f,0.f,0.f,0.f,0.f};
                v8f a1 = {0.f,0.f,0.f,0.f,0.f,0.f,0.f,0.f};
                gemm_tile2(Al + 320, 576, Wt, 128, nt2 * 32, 4, lane, a0, a1);
                store_C(&Gb[cross][0], 512, k * 128 + nt2 * 32,      a0, lane);
                store_C(&Gb[cross][0], 512, k * 128 + nt2 * 32 + 16, a1, lane);
            } else {
                int idx = job - 32, gm = idx >> 2, nt2 = idx & 3;
                const __bf16 *Asrc, *Wt; const float* bias; int st;
                if (gm == 0)      { Asrc = Av + 64;  st = 320; Wt = P.Wv2l1t; bias = P.bl1; }
                else if (gm == 1) { Asrc = Al + 320; st = 576; Wt = P.Wl2v1t; bias = P.bv1; }
                else if (gm == 2) { Asrc = Aa + 96;  st = 352; Wt = P.Wv2l2t; bias = P.bl2; }
                else              { Asrc = Al + 320; st = 576; Wt = P.Wl2v2t; bias = P.bv2; }
                v8f a0 = {0.f,0.f,0.f,0.f,0.f,0.f,0.f,0.f};
                v8f a1 = {0.f,0.f,0.f,0.f,0.f,0.f,0.f,0.f};
                gemm_tile2(Asrc, st, Wt, 128, nt2 * 32, 4, lane, a0, a1);
                float b0 = bias[nt2 * 32 + (lane & 15)];
                float b1 = bias[nt2 * 32 + 16 + (lane & 15)];
#pragma unroll
                for (int r = 0; r < 8; ++r) { a0[r] += b0; a1[r] += b1; }
                store_C(&Gb[2][0], 512, gm * 128 + nt2 * 32,      a0, lane);
                store_C(&Gb[2][0], 512, gm * 128 + nt2 * 32 + 16, a1, lane);
            }
        }
        __syncthreads();
        // ---- phase 5: bilinear correlation gates -----------------------
        if (tid < 128) {  // tid = cross*64 + r*4 + k
            int cross = tid >> 6, r = (tid >> 2) & 15, k = tid & 3;
            const float* hoth = cross ? &Hf[1][r * 128] : &Hf[2][r * 128];
            const float* u = &Gb[cross][r * 512 + k * 128];
            float d = 0.0f;
            for (int j = 0; j < 128; ++j) d += u[j] * hoth[j];
            Gg2[tid] = tanhf(d * 0.08838834764831845f);   // 1/sqrt(128)
        }
        __syncthreads();
        if (tid < 32) {
            int cross = tid >> 4, r = tid & 15;
            float m = Gg2[cross * 64 + r * 4] + Gg2[cross * 64 + r * 4 + 1]
                    + Gg2[cross * 64 + r * 4 + 2] + Gg2[cross * 64 + r * 4 + 3];
            Gg[tid] = sigm(m * 0.25f);
        }
        __syncthreads();
        // ---- phase 6: residual mixes -> h_cat (bf16) -------------------
        for (int i = tid; i < 2048; i += 256) {
            int r = i >> 7, j = i & 127;
            float g1 = Gg[r], g2 = Gg[16 + r];
            float hl = Hf[0][i], ha = Hf[1][i], hv = Hf[2][i];
            float l1 = hl + g1 * tanhf(Gb[2][r * 512 + j]);
            float vh = hv + g1 * tanhf(Gb[2][r * 512 + 128 + j]);
            float l2 = hl + g2 * tanhf(Gb[2][r * 512 + 256 + j]);
            float ah = ha + g2 * tanhf(Gb[2][r * 512 + 384 + j]);
            float lh = 0.7f * l1 + 0.3f * l2;
            Hcat[r * 384 + j]       = (__bf16)lh;
            Hcat[r * 384 + 128 + j] = (__bf16)vh;
            Hcat[r * 384 + 256 + j] = (__bf16)ah;
        }
        __syncthreads();
        // ---- phase 7: z_hat/gamma layer 1 (384 -> 256), 16 jobs --------
        for (int job = uwv; job < 16; job += 8) {
            int which = job >> 3, nt2 = job & 7;
            const __bf16* Wt = which ? P.gW1t : P.zW1t;
            const float* bias = which ? P.gb1 : P.zb1;
            v8f a0 = {0.f,0.f,0.f,0.f,0.f,0.f,0.f,0.f};
            v8f a1 = {0.f,0.f,0.f,0.f,0.f,0.f,0.f,0.f};
            gemm_tile2(Hcat, 384, Wt, 384, nt2 * 32, 12, lane, a0, a1);
            float b0 = bias[nt2 * 32 + (lane & 15)];
            float b1 = bias[nt2 * 32 + 16 + (lane & 15)];
#pragma unroll
            for (int r = 0; r < 8; ++r) { a0[r] += b0; a1[r] += b1; }
            store_C(&Gb[which][0], 256, nt2 * 32,      a0, lane);
            store_C(&Gb[which][0], 256, nt2 * 32 + 16, a1, lane);
        }
        __syncthreads();
        // ---- phase 8: fp32 -> bf16 for layer-2 A operand ---------------
        for (int i = tid; i < 2 * 4096; i += 256) {
            int w = i >> 12, idx = i & 4095;
            Za[w][idx] = (__bf16)Gb[w][idx];
        }
        __syncthreads();
        // ---- phase 9: layer 2 (256 -> 128), 8 jobs ---------------------
        {
            int job = uwv;
            int which = job >> 2, nt2 = job & 3;
            const __bf16* Wt = which ? P.gW2t : P.zW2t;
            const float* bias = which ? P.gb2 : P.zb2;
            v8f a0 = {0.f,0.f,0.f,0.f,0.f,0.f,0.f,0.f};
            v8f a1 = {0.f,0.f,0.f,0.f,0.f,0.f,0.f,0.f};
            gemm_tile2(Za[which], 256, Wt, 256, nt2 * 32, 8, lane, a0, a1);
            float b0 = bias[nt2 * 32 + (lane & 15)];
            float b1 = bias[nt2 * 32 + 16 + (lane & 15)];
#pragma unroll
            for (int r = 0; r < 8; ++r) { a0[r] += b0; a1[r] += b1; }
            store_C(&Gb[2][which * 2048], 128, nt2 * 32,      a0, lane);
            store_C(&Gb[2][which * 2048], 128, nt2 * 32 + 16, a1, lane);
        }
        __syncthreads();
        // ---- phase 10: z = sigmoid(gamma) * tanh(z_hat) ----------------
        for (int i = tid; i < 2048; i += 256) {
            int r = i >> 7, j = i & 127;
            float z = sigm(Gb[2][2048 + i]) * tanhf(Gb[2][i]);
            __bf16 zb = (__bf16)z;
            Al[r * 576 + 448 + j] = zb;
            Aa[r * 352 + 224 + j] = zb;
            Av[r * 320 + 192 + j] = zb;
        }
        __syncthreads();
    }

    // ---- prediction head: relu(z @ W1 + b1) @ W2 + b2 ------------------
    if (uwv < 4) {
        int nt2 = uwv;  // 4 waves x 2 tiles = 8 N-tiles
        v8f a0 = {0.f,0.f,0.f,0.f,0.f,0.f,0.f,0.f};
        v8f a1 = {0.f,0.f,0.f,0.f,0.f,0.f,0.f,0.f};
        gemm_tile2(Al + 448, 576, P.pW1t, 128, nt2 * 32, 4, lane, a0, a1);
        float b0 = P.pb1[nt2 * 32 + (lane & 15)];
        float b1 = P.pb1[nt2 * 32 + 16 + (lane & 15)];
#pragma unroll
        for (int r = 0; r < 8; ++r) {
            a0[r] = fmaxf(a0[r] + b0, 0.0f);
            a1[r] = fmaxf(a1[r] + b1, 0.0f);
        }
        store_C(&Gb[0][0], 128, nt2 * 32,      a0, lane);
        store_C(&Gb[0][0], 128, nt2 * 32 + 16, a1, lane);
    }
    __syncthreads();
    for (int i = tid; i < 16 * 3; i += 256) {
        int r = i / 3, o = i - r * 3;
        float s = P.pb2[o];
        for (int j = 0; j < 128; ++j) s += Gb[0][r * 128 + j] * P.pW2[j * 3 + o];
        P.out[(size_t)(r0 + r) * 3 + o] = s;
    }
}

// ---------------------------------------------------------------------------
// Weight prep: zero + transpose fp32 [K][N] -> padded bf16 [N][Kpad]
// ---------------------------------------------------------------------------
__global__ void zero_bf16(__bf16* p, int n) {
    int i = blockIdx.x * 256 + threadIdx.x;
    if (i < n) p[i] = (__bf16)0.0f;
}

__global__ void transpose_bf16_k(const float* W, __bf16* out, int K, int N, int Kpad, int koff) {
    int i = blockIdx.x * 256 + threadIdx.x;
    if (i >= K * N) return;
    int k = i / N, n = i - k * N;
    out[(size_t)n * Kpad + koff + k] = (__bf16)W[(size_t)k * N + n];
}

extern "C" void kernel_launch(void* const* d_in, const int* in_sizes, int n_in,
                              void* d_out, int out_size, void* d_ws, size_t ws_size,
                              hipStream_t stream) {
    // Input order: x_l, x_v, x_a, IsLastBatch, then params (insertion order).
    const float* x_l  = (const float*)d_in[0];
    const float* x_v  = (const float*)d_in[1];
    const float* x_a  = (const float*)d_in[2];
    const float* gvW  = (const float*)d_in[4];
    const float* gvB  = (const float*)d_in[5];
    const float* gaW  = (const float*)d_in[6];
    const float* gaB  = (const float*)d_in[7];
    const float* WxL  = (const float*)d_in[8];
    const float* WhL  = (const float*)d_in[9];
    const float* WzL  = (const float*)d_in[10];
    const float* bL   = (const float*)d_in[11];
    const float* WxA  = (const float*)d_in[12];
    const float* WhA  = (const float*)d_in[13];
    const float* WzA  = (const float*)d_in[14];
    const float* bA   = (const float*)d_in[15];
    const float* WxV  = (const float*)d_in[16];
    const float* WhV  = (const float*)d_in[17];
    const float* WzV  = (const float*)d_in[18];
    const float* bV   = (const float*)d_in[19];
    const float* Wk1  = (const float*)d_in[20];
    const float* Wv2l1= (const float*)d_in[21];
    const float* bl1  = (const float*)d_in[22];
    const float* Wl2v1= (const float*)d_in[23];
    const float* bv1  = (const float*)d_in[24];
    const float* Wk2  = (const float*)d_in[25];
    const float* Wv2l2= (const float*)d_in[26];
    const float* bl2  = (const float*)d_in[27];
    const float* Wl2v2= (const float*)d_in[28];
    const float* bv2  = (const float*)d_in[29];
    const float* zW1  = (const float*)d_in[30];
    const float* zb1  = (const float*)d_in[31];
    const float* zW2  = (const float*)d_in[32];
    const float* zb2  = (const float*)d_in[33];
    const float* gW1  = (const float*)d_in[34];
    const float* gb1  = (const float*)d_in[35];
    const float* gW2  = (const float*)d_in[36];
    const float* gb2  = (const float*)d_in[37];
    const float* pW1  = (const float*)d_in[38];
    const float* pb1  = (const float*)d_in[39];
    const float* pW2  = (const float*)d_in[40];
    const float* pb2  = (const float*)d_in[41];

    // Workspace layout (bf16 elements)
    __bf16* ws = (__bf16*)d_ws;
    size_t off = 0;
    __bf16* WcatL  = ws + off; off += (size_t)512 * 576;
    __bf16* WcatA  = ws + off; off += (size_t)512 * 352;
    __bf16* WcatV  = ws + off; off += (size_t)512 * 320;
    __bf16* Wk1t   = ws + off; off += (size_t)4 * 128 * 128;
    __bf16* Wk2t   = ws + off; off += (size_t)4 * 128 * 128;
    __bf16* Wv2l1t = ws + off; off += (size_t)128 * 128;
    __bf16* Wl2v1t = ws + off; off += (size_t)128 * 128;
    __bf16* Wv2l2t = ws + off; off += (size_t)128 * 128;
    __bf16* Wl2v2t = ws + off; off += (size_t)128 * 128;
    __bf16* zW1t   = ws + off; off += (size_t)256 * 384;
    __bf16* zW2t   = ws + off; off += (size_t)128 * 256;
    __bf16* gW1t   = ws + off; off += (size_t)256 * 384;
    __bf16* gW2t   = ws + off; off += (size_t)128 * 256;
    __bf16* pW1t   = ws + off; off += (size_t)128 * 128;
    int total = (int)off;

    zero_bf16<<<(total + 255) / 256, 256, 0, stream>>>(ws, total);

    auto TR = [&](const float* W, __bf16* o, int K, int N, int Kpad, int koff) {
        transpose_bf16_k<<<(K * N + 255) / 256, 256, 0, stream>>>(W, o, K, N, Kpad, koff);
    };
    TR(WxL, WcatL, 300, 512, 576, 0);
    TR(WhL, WcatL, 128, 512, 576, 320);
    TR(WzL, WcatL, 128, 512, 576, 448);
    TR(WxA, WcatA, 74,  512, 352, 0);
    TR(WhA, WcatA, 128, 512, 352, 96);
    TR(WzA, WcatA, 128, 512, 352, 224);
    TR(WxV, WcatV, 47,  512, 320, 0);
    TR(WhV, WcatV, 128, 512, 320, 64);
    TR(WzV, WcatV, 128, 512, 320, 192);
    for (int k = 0; k < 4; ++k) {
        TR(Wk1 + (size_t)k * 16384, Wk1t + (size_t)k * 16384, 128, 128, 128, 0);
        TR(Wk2 + (size_t)k * 16384, Wk2t + (size_t)k * 16384, 128, 128, 128, 0);
    }
    TR(Wv2l1, Wv2l1t, 128, 128, 128, 0);
    TR(Wl2v1, Wl2v1t, 128, 128, 128, 0);
    TR(Wv2l2, Wv2l2t, 128, 128, 128, 0);
    TR(Wl2v2, Wl2v2t, 128, 128, 128, 0);
    TR(zW1, zW1t, 384, 256, 384, 0);
    TR(zW2, zW2t, 256, 128, 256, 0);
    TR(gW1, gW1t, 384, 256, 384, 0);
    TR(gW2, gW2t, 256, 128, 256, 0);
    TR(pW1, pW1t, 128, 128, 128, 0);

    KParams P;
    P.x_l = x_l; P.x_v = x_v; P.x_a = x_a;
    P.gvW = gvW; P.gvB = gvB; P.gaW = gaW; P.gaB = gaB;
    P.bL = bL; P.bA = bA; P.bV = bV;
    P.bl1 = bl1; P.bv1 = bv1; P.bl2 = bl2; P.bv2 = bv2;
    P.zb1 = zb1; P.zb2 = zb2; P.gb1 = gb1; P.gb2 = gb2;
    P.pb1 = pb1; P.pW2 = pW2; P.pb2 = pb2;
    P.WcatL = WcatL; P.WcatA = WcatA; P.WcatV = WcatV;
    P.Wk1t = Wk1t; P.Wk2t = Wk2t;
    P.Wv2l1t = Wv2l1t; P.Wl2v1t = Wl2v1t; P.Wv2l2t = Wv2l2t; P.Wl2v2t = Wl2v2t;
    P.zW1t = zW1t; P.zW2t = zW2t; P.gW1t = gW1t; P.gW2t = gW2t;
    P.pW1t = pW1t;
    P.out = (float*)d_out;

    fuse_kernel<<<128, 256, 0, stream>>>(P);
}
